// GatedAttentionBlock_592705486936
// MI455X (gfx1250) — matrix operand order repaired and checked by reading, not compiled, not run on hardware
//
#include <hip/hip_runtime.h>
#include <math.h>

// ---------------------------------------------------------------------------
// GatedAttentionBlock on MI455X (gfx1250).
// bf16 WMMA everywhere; operands pre-converted (and weights pre-transposed to
// [N][K]) to bf16 once per call so GEMM tile staging is pure 16B copies done
// with GLOBAL_LOAD_ASYNC_TO_LDS_B128, double-buffered against WMMA compute.
// ---------------------------------------------------------------------------

constexpr int Bc = 2, Sc = 2048, Dc = 2048, Hc = 16, HDc = 128;
constexpr int Mrows = Bc * Sc;            // 4096 tokens
constexpr float RMS_EPS = 1.1920929e-07f;

typedef __attribute__((ext_vector_type(16))) __bf16 v16bf;
typedef __attribute__((ext_vector_type(8)))  float  v8f;

union BF16Frag {
  v16bf v;
  uint4 q[2];
  unsigned short s[16];
};

__device__ __forceinline__ unsigned short f2bf(float f) {
  union { float f; unsigned u; } c; c.f = f;
  unsigned u = c.u + 0x7FFFu + ((c.u >> 16) & 1u);   // RNE truncate to bf16
  return (unsigned short)(u >> 16);
}

__device__ __forceinline__ void store_bf4(unsigned short* p, float4 v) {
  union { unsigned short s[4]; uint2 d; } t;
  t.s[0] = f2bf(v.x); t.s[1] = f2bf(v.y); t.s[2] = f2bf(v.z); t.s[3] = f2bf(v.w);
  *(uint2*)p = t.d;
}

__device__ __forceinline__ v8f wmma_bf16(const BF16Frag& a, const BF16Frag& b, v8f c) {
  return __builtin_amdgcn_wmma_f32_16x16x32_bf16(
      false, a.v, false, b.v, (short)0, c, false, false);
}

// --- CDNA5 async global->LDS copy (16B per lane), ASYNCcnt-tracked ----------
__device__ __forceinline__ void async_copy_b128(unsigned lds_off, const void* gptr) {
  asm volatile("global_load_async_to_lds_b128 %0, %1, off"
               :: "v"(lds_off), "v"((unsigned long long)(size_t)gptr)
               : "memory");
}
__device__ __forceinline__ void wait_async0() {
  asm volatile("s_wait_asynccnt 0" ::: "memory");
}
// Low 32 bits of a flat pointer into the LDS aperture = LDS byte offset.
__device__ __forceinline__ unsigned lds_off32(const void* p) {
  return (unsigned)(size_t)p;
}

// ---------------------------------------------------------------------------
// RMSNorm: one row per block, bf16 output (GEMM A operand).
// ---------------------------------------------------------------------------
__global__ __launch_bounds__(256)
void rmsnorm_bf16_kernel(const float* __restrict__ x, unsigned short* __restrict__ y,
                         int ncols) {
  __shared__ float red[8];
  const int row = blockIdx.x, tid = threadIdx.x;
  const float* xr = x + (size_t)row * ncols;
  float s = 0.f;
  for (int i = tid; i < ncols; i += 256) { float v = xr[i]; s += v * v; }
  s += __shfl_xor(s, 1);  s += __shfl_xor(s, 2);
  s += __shfl_xor(s, 4);  s += __shfl_xor(s, 8);  s += __shfl_xor(s, 16);
  if ((tid & 31) == 0) red[tid >> 5] = s;
  __syncthreads();
  float tot = 0.f;
  #pragma unroll
  for (int w = 0; w < 8; ++w) tot += red[w];
  const float r = rsqrtf(tot / (float)ncols + RMS_EPS);
  unsigned short* yr = y + (size_t)row * ncols;
  for (int i = tid; i < ncols; i += 256) yr[i] = f2bf(xr[i] * r);
}

// ---------------------------------------------------------------------------
// Weight convert+transpose: W[K][N] fp32 -> Wt[N][K] bf16 (32x32 LDS tiles).
// One-time ~430MB pass so every GEMM streams half the bytes and both GEMM
// operands share the contiguous-K layout required by async tile staging.
// ---------------------------------------------------------------------------
__global__ __launch_bounds__(256)
void convert_transpose_kernel(const float* __restrict__ W,
                              unsigned short* __restrict__ Wt, int K, int N) {
  __shared__ unsigned short t[32][33];
  const int tid = threadIdx.x;
  const int k0 = blockIdx.y * 32, n0 = blockIdx.x * 32;
  #pragma unroll
  for (int l = 0; l < 4; ++l) {
    int idx = tid + l * 256;
    int i = idx >> 5, j = idx & 31;
    t[i][j] = f2bf(W[(size_t)(k0 + i) * N + n0 + j]);
  }
  __syncthreads();
  #pragma unroll
  for (int l = 0; l < 4; ++l) {
    int idx = tid + l * 256;
    int n = idx >> 5, k = idx & 31;
    Wt[(size_t)(n0 + n) * K + k0 + k] = t[k][n];
  }
}

// ---------------------------------------------------------------------------
// Householder product Q (128x128); thread j owns column j in registers.
// ---------------------------------------------------------------------------
__global__ __launch_bounds__(128)
void householder_kernel(const float* __restrict__ vs, float* __restrict__ Qout) {
  const int j = threadIdx.x;
  float col[128];
  #pragma unroll
  for (int i = 0; i < 128; ++i) col[i] = (i == j) ? 1.f : 0.f;
  for (int st = 0; st < 64; ++st) {
    const float* v = vs + st * 128;
    float dot = 0.f;
    #pragma unroll
    for (int i = 0; i < 128; ++i) dot += v[i] * v[i];
    const float coef = 2.f / (dot + 1e-8f);
    float w = 0.f;
    #pragma unroll
    for (int i = 0; i < 128; ++i) w += v[i] * col[i];
    const float cw = coef * w;
    #pragma unroll
    for (int i = 0; i < 128; ++i) col[i] -= cw * v[i];
  }
  #pragma unroll
  for (int i = 0; i < 128; ++i) Qout[i * 128 + j] = col[i];
}

// ---------------------------------------------------------------------------
// Per-token q/k transform: t -> rot(t @ Q^T) @ Q, head-major [B,H,S,HD] fp32.
// ---------------------------------------------------------------------------
__global__ __launch_bounds__(128)
void qk_transform_kernel(const float* __restrict__ qkv, const float* __restrict__ Qm,
                         const float* __restrict__ inv_freq,
                         const float* __restrict__ rope_pos,
                         const float* __restrict__ scale,
                         float* __restrict__ qr, float* __restrict__ kr) {
  __shared__ float tq[128], tk[128], yq[128], yk[128], rq[128], rk[128];
  const int j = threadIdx.x;
  const int s = blockIdx.x, b = blockIdx.y;
  const size_t rowbase = ((size_t)(b * Sc) + s) * (size_t)(3 * Dc);
  const float invsc = 1.f / scale[0];
  const float pos = rope_pos[s * 2 + ((j & 63) >> 5)];
  const float ang = pos * inv_freq[j & 31] * invsc;
  const float cs = __cosf(ang), sn = __sinf(ang);

  for (int h = 0; h < Hc; ++h) {
    tq[j] = qkv[rowbase + h * HDc + j];
    tk[j] = qkv[rowbase + Dc + h * HDc + j];
    __syncthreads();
    float aq = 0.f, ak = 0.f;                      // y = t @ Q^T
    const float* Qrow = Qm + j * HDc;
    for (int d = 0; d < HDc; ++d) {
      float qv = Qrow[d];
      aq += tq[d] * qv;  ak += tk[d] * qv;
    }
    yq[j] = aq; yk[j] = ak;
    __syncthreads();
    const int part = j ^ 64;
    const float oq = (j < 64) ? -yq[part] : yq[part];
    const float ok = (j < 64) ? -yk[part] : yk[part];
    rq[j] = yq[j] * cs + oq * sn;
    rk[j] = yk[j] * cs + ok * sn;
    __syncthreads();
    float zq = 0.f, zk = 0.f;                      // z = yr @ Q
    for (int d = 0; d < HDc; ++d) {
      float qv = Qm[d * HDc + j];
      zq += rq[d] * qv;  zk += rk[d] * qv;
    }
    const size_t o = (((size_t)(b * Hc + h)) * Sc + s) * HDc + j;
    qr[o] = zq;  kr[o] = zk;
    __syncthreads();
  }
}

// ---------------------------------------------------------------------------
// Async double-buffered bf16 WMMA GEMM core.
// Block tile 128x128, BK=32, 256 threads = 8 waves (4x2), wave tile 32x64.
// Both operands are bf16 with contiguous K (weights pre-transposed [N][K]),
// staged into padded pitch-40 LDS via GLOBAL_LOAD_ASYNC_TO_LDS_B128 while the
// previous tile is consumed by v_wmma_f32_16x16x32_bf16.
// ---------------------------------------------------------------------------
__device__ __forceinline__ void issue_tile(const unsigned short* __restrict__ G,
                                           int ld, int row0, int k0,
                                           unsigned lds_base, int tid) {
  // 128 rows x 32 bf16 = 512 chunks of 16B, 2 per thread.
  #pragma unroll
  for (int l = 0; l < 2; ++l) {
    int idx = tid + l * 256;
    int r = idx >> 2, seg = idx & 3;
    const unsigned short* g = G + (size_t)(row0 + r) * ld + k0 + seg * 8;
    async_copy_b128(lds_base + (unsigned)(r * 80 + seg * 16), g);
  }
}

__device__ __forceinline__ void gemm_pass_async(
    const unsigned short* __restrict__ Abf, const unsigned short* __restrict__ Wt,
    int K, int lda, int ldw, int m0, int n0,
    unsigned short (*Asm)[128 * 40], unsigned short (*Bsm)[128 * 40],
    v8f acc[2][4]) {
  const int tid = threadIdx.x;
  const int lane = tid & 31, wv = tid >> 5;
  const int wm = wv >> 1, wn = wv & 1;
  const int hi = lane >> 4, l15 = lane & 15;
  const unsigned aoff[2] = { lds_off32(Asm[0]), lds_off32(Asm[1]) };
  const unsigned boff[2] = { lds_off32(Bsm[0]), lds_off32(Bsm[1]) };
  const int nk = K >> 5;

  issue_tile(Abf, lda, m0, 0, aoff[0], tid);
  issue_tile(Wt,  ldw, n0, 0, boff[0], tid);
  wait_async0();
  __syncthreads();

  for (int kt = 0; kt < nk; ++kt) {
    const int cur = kt & 1;
    if (kt + 1 < nk) {                       // prefetch next tile into other buf
      issue_tile(Abf, lda, m0, (kt + 1) * 32, aoff[cur ^ 1], tid);
      issue_tile(Wt,  ldw, n0, (kt + 1) * 32, boff[cur ^ 1], tid);
    }
    BF16Frag a[2], b[4];
    #pragma unroll
    for (int mi = 0; mi < 2; ++mi) {
      const unsigned short* p = &Asm[cur][(wm * 32 + mi * 16 + l15) * 40 + hi * 8];
      a[mi].q[0] = *(const uint4*)p;
      a[mi].q[1] = *(const uint4*)(p + 16);
    }
    #pragma unroll
    for (int ni = 0; ni < 4; ++ni) {
      const unsigned short* p = &Bsm[cur][(wn * 64 + ni * 16 + l15) * 40 + hi * 8];
      b[ni].q[0] = *(const uint4*)p;
      b[ni].q[1] = *(const uint4*)(p + 16);
    }
    #pragma unroll
    for (int mi = 0; mi < 2; ++mi)
      #pragma unroll
      for (int ni = 0; ni < 4; ++ni)
        acc[mi][ni] = wmma_bf16(a[mi], b[ni], acc[mi][ni]);
    wait_async0();                           // my async loads for next tile done
    __syncthreads();                         // everyone's done (loads + LDS reads)
  }
}

enum { EPI_NONE = 0, EPI_GATE = 1, EPI_RESID = 2 };

template <int EPI>
__global__ __launch_bounds__(256)
void gemm_bf16_kernel(const unsigned short* __restrict__ A,
                      const unsigned short* __restrict__ Wt,
                      float* __restrict__ out, unsigned short* __restrict__ out_bf,
                      int M, int N, int K, int lda, int ldw, int ldo,
                      const float* __restrict__ bias,
                      const float* __restrict__ proj,
                      const float* __restrict__ resid) {
  __shared__ unsigned short Asm[2][128 * 40];
  __shared__ unsigned short Bsm[2][128 * 40];
  const int tid = threadIdx.x;
  const int lane = tid & 31, wv = tid >> 5;
  const int wm = wv >> 1, wn = wv & 1;
  const int hi = lane >> 4, l15 = lane & 15;
  const int m0 = blockIdx.y * 128, n0 = blockIdx.x * 128;

  v8f acc[2][4];
  #pragma unroll
  for (int i = 0; i < 2; ++i)
    #pragma unroll
    for (int j = 0; j < 4; ++j) acc[i][j] = {};

  gemm_pass_async(A, Wt, K, lda, ldw, m0, n0, Asm, Bsm, acc);

  #pragma unroll
  for (int mi = 0; mi < 2; ++mi)
    #pragma unroll
    for (int ni = 0; ni < 4; ++ni)
      #pragma unroll
      for (int r = 0; r < 8; ++r) {
        const int gm = m0 + wm * 32 + mi * 16 + r + hi * 8;
        const int gn = n0 + wn * 64 + ni * 16 + l15;
        const size_t oidx = (size_t)gm * ldo + gn;
        const float v = acc[mi][ni][r];
        float res;
        if (EPI == EPI_NONE) {
          res = v;
        } else if (EPI == EPI_GATE) {
          const float g = 1.f / (1.f + __expf(-(v + bias[gn])));
          res = resid[oidx] + proj[oidx] * g;
        } else {  // EPI_RESID
          res = resid[oidx] + v;
        }
        out[oidx] = res;
        if (out_bf) out_bf[oidx] = f2bf(res);
      }
}

// Fused SwiGLU GEMM: two K-passes over w12t row halves, h = silu(x1)*x2 (bf16).
__global__ __launch_bounds__(256)
void swiglu_gemm_kernel(const unsigned short* __restrict__ A,
                        const unsigned short* __restrict__ Wt,
                        unsigned short* __restrict__ out,
                        int K, int lda, int ldw, int half_off, int ldo) {
  __shared__ unsigned short Asm[2][128 * 40];
  __shared__ unsigned short Bsm[2][128 * 40];
  const int tid = threadIdx.x;
  const int lane = tid & 31, wv = tid >> 5;
  const int wm = wv >> 1, wn = wv & 1;
  const int hi = lane >> 4, l15 = lane & 15;
  const int m0 = blockIdx.y * 128, n0 = blockIdx.x * 128;

  v8f a1[2][4], a2[2][4];
  #pragma unroll
  for (int i = 0; i < 2; ++i)
    #pragma unroll
    for (int j = 0; j < 4; ++j) { a1[i][j] = {}; a2[i][j] = {}; }

  gemm_pass_async(A, Wt, K, lda, ldw, m0, n0, Asm, Bsm, a1);
  gemm_pass_async(A, Wt, K, lda, ldw, m0, n0 + half_off, Asm, Bsm, a2);

  #pragma unroll
  for (int mi = 0; mi < 2; ++mi)
    #pragma unroll
    for (int ni = 0; ni < 4; ++ni)
      #pragma unroll
      for (int r = 0; r < 8; ++r) {
        const int gm = m0 + wm * 32 + mi * 16 + r + hi * 8;
        const int gn = n0 + wn * 64 + ni * 16 + l15;
        const float x1 = a1[mi][ni][r];
        const float x2 = a2[mi][ni][r];
        const float silu = x1 / (1.f + __expf(-x1));
        out[(size_t)gm * ldo + gn] = f2bf(silu * x2);
      }
}

// ---------------------------------------------------------------------------
// Flash-style causal attention (unchanged WMMA core); output stored bf16 since
// its only consumer is the out-projection GEMM.
// ---------------------------------------------------------------------------
__global__ __launch_bounds__(256)
void attention_kernel(const float* __restrict__ Qr, const float* __restrict__ Kr,
                      const float* __restrict__ qkv,
                      unsigned short* __restrict__ ctx_bf) {
  __shared__ unsigned short Ksm[64 * 136];    // [krow][d]
  __shared__ unsigned short Vsm[128 * 72];    // [d][krow]
  __shared__ unsigned short Psm[8 * 16 * 72]; // per-wave 16x64 P tile
  const int tid = threadIdx.x, lane = tid & 31, wv = tid >> 5;
  const int hi = lane >> 4, l15 = lane & 15;
  const int qt = blockIdx.x, bh = blockIdx.y;
  const int b = bh / Hc, h = bh % Hc;
  const int qbase = qt * 128;
  const float* qhead = Qr + (size_t)bh * Sc * HDc;
  const float* khead = Kr + (size_t)bh * Sc * HDc;
  const float sscale = 0.08838834764831845f;  // 1/sqrt(128)

  BF16Frag qf[4];
  {
    const float* qp = qhead + (size_t)(qbase + wv * 16 + l15) * HDc;
    #pragma unroll
    for (int ks = 0; ks < 4; ++ks) {
      const int d0 = ks * 32 + hi * 8;
      float4 u0 = *(const float4*)(qp + d0);
      float4 u1 = *(const float4*)(qp + d0 + 4);
      float4 u2 = *(const float4*)(qp + d0 + 16);
      float4 u3 = *(const float4*)(qp + d0 + 20);
      BF16Frag t;
      t.s[0] = f2bf(u0.x); t.s[1] = f2bf(u0.y); t.s[2]  = f2bf(u0.z); t.s[3]  = f2bf(u0.w);
      t.s[4] = f2bf(u1.x); t.s[5] = f2bf(u1.y); t.s[6]  = f2bf(u1.z); t.s[7]  = f2bf(u1.w);
      t.s[8] = f2bf(u2.x); t.s[9] = f2bf(u2.y); t.s[10] = f2bf(u2.z); t.s[11] = f2bf(u2.w);
      t.s[12]= f2bf(u3.x); t.s[13]= f2bf(u3.y); t.s[14] = f2bf(u3.z); t.s[15] = f2bf(u3.w);
      qf[ks] = t;
    }
  }

  v8f cacc[8];
  #pragma unroll
  for (int n = 0; n < 8; ++n) cacc[n] = {};
  float mstate[8], lstate[8];
  #pragma unroll
  for (int r = 0; r < 8; ++r) { mstate[r] = -1e30f; lstate[r] = 0.f; }

  const int nkt = qt * 2 + 2;
  for (int kt = 0; kt < nkt; ++kt) {
    const int kbase = kt * 64;
    __syncthreads();
    #pragma unroll
    for (int l = 0; l < 8; ++l) {               // K tile 64x128
      int idx = tid + l * 256;
      int r = idx >> 5, d = (idx & 31) * 4;
      float4 v = *(const float4*)(khead + (size_t)(kbase + r) * HDc + d);
      store_bf4(&Ksm[r * 136 + d], v);
    }
    #pragma unroll
    for (int l = 0; l < 8; ++l) {               // V tile (from qkv), transposed
      int idx = tid + l * 256;
      int r = idx >> 5, d = (idx & 31) * 4;
      float4 v = *(const float4*)(qkv + ((size_t)(b * Sc) + kbase + r) * (3 * Dc)
                                  + 2 * Dc + h * HDc + d);
      Vsm[(d + 0) * 72 + r] = f2bf(v.x);
      Vsm[(d + 1) * 72 + r] = f2bf(v.y);
      Vsm[(d + 2) * 72 + r] = f2bf(v.z);
      Vsm[(d + 3) * 72 + r] = f2bf(v.w);
    }
    __syncthreads();

    v8f sacc[4];
    #pragma unroll
    for (int ct = 0; ct < 4; ++ct) sacc[ct] = {};
    #pragma unroll
    for (int ks = 0; ks < 4; ++ks)
      #pragma unroll
      for (int ct = 0; ct < 4; ++ct) {
        BF16Frag kb;
        const unsigned short* p = &Ksm[(ct * 16 + l15) * 136 + ks * 32 + hi * 8];
        kb.q[0] = *(const uint4*)p;
        kb.q[1] = *(const uint4*)(p + 16);
        sacc[ct] = wmma_bf16(qf[ks], kb, sacc[ct]);
      }

    #pragma unroll
    for (int r = 0; r < 8; ++r) {
      const int grow = qbase + wv * 16 + r + hi * 8;
      float mx = -1e30f;
      #pragma unroll
      for (int ct = 0; ct < 4; ++ct) {
        const int gcol = kbase + ct * 16 + l15;
        float sv = sacc[ct][r] * sscale;
        sv = (gcol <= grow) ? sv : -1e30f;
        sacc[ct][r] = sv;
        mx = fmaxf(mx, sv);
      }
      mx = fmaxf(mx, __shfl_xor(mx, 1));
      mx = fmaxf(mx, __shfl_xor(mx, 2));
      mx = fmaxf(mx, __shfl_xor(mx, 4));
      mx = fmaxf(mx, __shfl_xor(mx, 8));
      const float mnew = fmaxf(mstate[r], mx);
      const float alpha = __expf(mstate[r] - mnew);
      float rowsum = 0.f;
      #pragma unroll
      for (int ct = 0; ct < 4; ++ct) {
        const float pv = __expf(sacc[ct][r] - mnew);
        rowsum += pv;
        Psm[wv * 1152 + (r + hi * 8) * 72 + ct * 16 + l15] = f2bf(pv);
      }
      rowsum += __shfl_xor(rowsum, 1);
      rowsum += __shfl_xor(rowsum, 2);
      rowsum += __shfl_xor(rowsum, 4);
      rowsum += __shfl_xor(rowsum, 8);
      lstate[r] = lstate[r] * alpha + rowsum;
      mstate[r] = mnew;
      #pragma unroll
      for (int n = 0; n < 8; ++n) cacc[n][r] *= alpha;
    }

    #pragma unroll
    for (int kk = 0; kk < 2; ++kk) {
      BF16Frag pf;
      const unsigned short* pp = &Psm[wv * 1152 + l15 * 72 + kk * 32 + hi * 8];
      pf.q[0] = *(const uint4*)pp;
      pf.q[1] = *(const uint4*)(pp + 16);
      #pragma unroll
      for (int n = 0; n < 8; ++n) {
        BF16Frag vb;
        const unsigned short* vp = &Vsm[(n * 16 + l15) * 72 + kk * 32 + hi * 8];
        vb.q[0] = *(const uint4*)vp;
        vb.q[1] = *(const uint4*)(vp + 16);
        cacc[n] = wmma_bf16(pf, vb, cacc[n]);
      }
    }
  }

  #pragma unroll
  for (int r = 0; r < 8; ++r) {
    const float inv = 1.f / lstate[r];
    const int grow = qbase + wv * 16 + r + hi * 8;
    #pragma unroll
    for (int n = 0; n < 8; ++n) {
      const int d = n * 16 + l15;
      ctx_bf[((size_t)(b * Sc) + grow) * Dc + h * HDc + d] = f2bf(cacc[n][r] * inv);
    }
  }
}

// ---------------------------------------------------------------------------
// Launcher. Workspace (~428 MB): bf16 activations, fp32 intermediates, and
// one-time bf16 [N][K]-transposed weight copies. qkv region reused for h.
// ---------------------------------------------------------------------------
extern "C" void kernel_launch(void* const* d_in, const int* in_sizes, int n_in,
                              void* d_out, int out_size, void* d_ws, size_t ws_size,
                              hipStream_t stream) {
  const float* x        = (const float*)d_in[0];
  const float* qkv_w    = (const float*)d_in[1];
  const float* out_w    = (const float*)d_in[2];
  const float* gate_w   = (const float*)d_in[3];
  const float* gate_b   = (const float*)d_in[4];
  const float* hh_vs    = (const float*)d_in[5];
  const float* w12      = (const float*)d_in[6];
  const float* w3       = (const float*)d_in[7];
  const float* inv_freq = (const float*)d_in[8];
  const float* rope_pos = (const float*)d_in[9];
  const float* scale    = (const float*)d_in[10];
  float* out = (float*)d_out;

  char* ws = (char*)d_ws;
  const size_t fM = (size_t)Mrows;
  size_t off = 0;
  auto carve = [&](size_t bytes) { size_t o = off; off += (bytes + 255) & ~(size_t)255; return o; };
  unsigned short* xnbf   = (unsigned short*)(ws + carve(fM * Dc * 2));
  float*          qkv    = (float*)         (ws + carve(fM * 6144 * 4)); // reused for hbf
  float*          qr     = (float*)         (ws + carve(fM * Dc * 4));
  float*          kr     = (float*)         (ws + carve(fM * Dc * 4));
  unsigned short* ctxbf  = (unsigned short*)(ws + carve(fM * Dc * 2));
  float*          proj   = (float*)         (ws + carve(fM * Dc * 4));
  unsigned short* projbf = (unsigned short*)(ws + carve(fM * Dc * 2));
  float*          x1     = (float*)         (ws + carve(fM * Dc * 4));
  float*          Qm     = (float*)         (ws + carve(128 * 128 * 4));
  unsigned short* wtqkv  = (unsigned short*)(ws + carve((size_t)Dc * 6144 * 2));
  unsigned short* wtout  = (unsigned short*)(ws + carve((size_t)Dc * Dc * 2));
  unsigned short* wtgate = (unsigned short*)(ws + carve((size_t)Dc * Dc * 2));
  unsigned short* wtw12  = (unsigned short*)(ws + carve((size_t)Dc * 16384 * 2));
  unsigned short* wtw3   = (unsigned short*)(ws + carve((size_t)8192 * Dc * 2));
  unsigned short* hbf    = (unsigned short*)qkv;   // reuse after attention
  (void)ws_size; (void)in_sizes; (void)n_in; (void)out_size;

  const dim3 blk256(256), blk128(128);

  // 0) one-time bf16 [N][K] weight copies
  convert_transpose_kernel<<<dim3(6144 / 32, Dc / 32),   blk256, 0, stream>>>(qkv_w,  wtqkv,  Dc,   6144);
  convert_transpose_kernel<<<dim3(Dc / 32, Dc / 32),     blk256, 0, stream>>>(out_w,  wtout,  Dc,   Dc);
  convert_transpose_kernel<<<dim3(Dc / 32, Dc / 32),     blk256, 0, stream>>>(gate_w, wtgate, Dc,   Dc);
  convert_transpose_kernel<<<dim3(16384 / 32, Dc / 32),  blk256, 0, stream>>>(w12,    wtw12,  Dc,   16384);
  convert_transpose_kernel<<<dim3(Dc / 32, 8192 / 32),   blk256, 0, stream>>>(w3,     wtw3,   8192, Dc);

  // 1) xn = rmsnorm(x)  (bf16)
  rmsnorm_bf16_kernel<<<Mrows, blk256, 0, stream>>>(x, xnbf, Dc);
  // 2) Householder Q
  householder_kernel<<<1, blk128, 0, stream>>>(hh_vs, Qm);
  // 3) qkv = xn @ qkv_w  [M,6144] fp32
  gemm_bf16_kernel<EPI_NONE><<<dim3(6144 / 128, Mrows / 128), blk256, 0, stream>>>(
      xnbf, wtqkv, qkv, nullptr, Mrows, 6144, Dc, Dc, Dc, 6144, nullptr, nullptr, nullptr);
  // 4) q,k -> rot(t Q^T) Q, head-major fp32
  qk_transform_kernel<<<dim3(Sc, Bc), blk128, 0, stream>>>(
      qkv, Qm, inv_freq, rope_pos, scale, qr, kr);
  // 5) causal attention -> ctx (bf16)
  attention_kernel<<<dim3(Sc / 128, Bc * Hc), blk256, 0, stream>>>(qr, kr, qkv, ctxbf);
  // 6) proj = ctx @ out_w  (fp32 + bf16 copy)
  gemm_bf16_kernel<EPI_NONE><<<dim3(Dc / 128, Mrows / 128), blk256, 0, stream>>>(
      ctxbf, wtout, proj, projbf, Mrows, Dc, Dc, Dc, Dc, Dc, nullptr, nullptr, nullptr);
  // 7) x1 = x + proj * sigmoid(proj @ gate_w + gate_b)
  gemm_bf16_kernel<EPI_GATE><<<dim3(Dc / 128, Mrows / 128), blk256, 0, stream>>>(
      projbf, wtgate, x1, nullptr, Mrows, Dc, Dc, Dc, Dc, Dc, gate_b, proj, x);
  // 8) xn = rmsnorm(x1)  (bf16)
  rmsnorm_bf16_kernel<<<Mrows, blk256, 0, stream>>>(x1, xnbf, Dc);
  // 9) h = silu(xn @ w12[:, :8192]) * (xn @ w12[:, 8192:])  (bf16)
  swiglu_gemm_kernel<<<dim3(8192 / 128, Mrows / 128), blk256, 0, stream>>>(
      xnbf, wtw12, hbf, Dc, Dc, Dc, 8192, 8192);
  // 10) out = x1 + h @ w3
  gemm_bf16_kernel<EPI_RESID><<<dim3(Dc / 128, Mrows / 128), blk256, 0, stream>>>(
      hbf, wtw3, out, nullptr, Mrows, Dc, 8192, 8192, 8192, Dc, nullptr, nullptr, x1);
}
